// Qwen3VLTextAttention_50044958933349
// MI455X (gfx1250) — matrix operand order repaired and checked by reading, not compiled, not run on hardware
//
#include <hip/hip_runtime.h>
#include <hip/hip_bf16.h>
#include <math.h>

// ---------------------------------------------------------------------------
// Qwen3 attention block on gfx1250 (CDNA5), bf16 WMMA + fp32 accumulate.
// B=2, S=2048, HID=2048, NH=16, NKV=4, HD=128
// GEMMs: block = 256x64 output strip (2 m-tiles per wave), B tiles staged in
// LDS with global_load_async_to_lds_b128 (ASYNCcnt double-buffering).
// ---------------------------------------------------------------------------

typedef __attribute__((ext_vector_type(16))) __bf16 v16bf;
typedef __attribute__((ext_vector_type(8)))  float  v8f;

#define WMMA_BF16(a, b, c) \
  __builtin_amdgcn_wmma_f32_16x16x32_bf16(false, (a), false, (b), (short)0, (c), false, false)

constexpr int Bc   = 2;
constexpr int Sc   = 2048;
constexpr int HIDc = 2048;
constexpr int NHc  = 16;
constexpr int NKVc = 4;
constexpr int HDc  = 128;
constexpr float EPSc = 1e-6f;

// ===========================================================================
// Fragment layout helpers (ISA 7.12.2, wave32):
//  A (16Mx32K bf16): lane l holds row M=l%16; elem i -> K = (i<8 ? i+8h : i+8+8h), h=l/16
//  B (32Kx16N bf16): lane l holds col N=l%16; elem i -> K = (i<8 ? i+8h : i+8+8h)
//  C/D (16x16 f32):  lane l holds col N=l%16; reg r -> M = r + 8h
// Packed tiles are 512 elements; lane's 16 elems are contiguous (32B aligned).
// ===========================================================================

__device__ __forceinline__ int a_kmap(int i, int half) {
  return (i < 8) ? (i + 8 * half) : (i + 8 + 8 * half);
}

// ---- pack row-major fp32 (M x K) into A-layout bf16 tiles ------------------
__global__ void pack_a_kernel(const float* __restrict__ src, __bf16* __restrict__ dst,
                              int M, int K, int total) {
  int t = blockIdx.x * blockDim.x + threadIdx.x;
  if (t >= total) return;
  int tile   = t >> 9;
  int within = t & 511;
  int lane = within >> 4;
  int i    = within & 15;
  int ktiles = K >> 5;
  int mt = tile / ktiles;
  int kt = tile - mt * ktiles;
  int half = lane >> 4;
  int row  = (mt << 4) + (lane & 15);
  int k    = (kt << 5) + a_kmap(i, half);
  dst[t] = (__bf16)src[(size_t)row * K + k];
}

// ---- pack row-major fp32 weight (K x N) into B-layout bf16 tiles -----------
__global__ void pack_b_kernel(const float* __restrict__ src, __bf16* __restrict__ dst,
                              int K, int N, int total) {
  int t = blockIdx.x * blockDim.x + threadIdx.x;
  if (t >= total) return;
  int tile   = t >> 9;
  int within = t & 511;
  int lane = within >> 4;
  int i    = within & 15;
  int ntiles = N >> 4;
  int kt = tile / ntiles;
  int nt = tile - kt * ntiles;
  int n = (nt << 4) + (lane & 15);
  int k = (kt << 5) + a_kmap(i, lane >> 4);
  dst[t] = (__bf16)src[(size_t)k * N + n];
}

// ---- block-cooperative GEMM: C(MxN,f32) = Ap(A-layout) @ Bp(B-layout) ------
// Block (8 waves) computes a 256x64 strip: wave w owns m-tiles (mblk*16+2w,+1)
// and all waves share 4 B fragments per k-step, staged in LDS by async copies.
// Per k-step/wave: 2 A loads, 4 shared B frags, 8 WMMAs.
__global__ __launch_bounds__(256) void gemm_bf16_kernel(
    const __bf16* __restrict__ Ap, const __bf16* __restrict__ Bp,
    float* __restrict__ C, int M, int K, int N) {
  __shared__ __attribute__((aligned(32))) __bf16 bsh[2][2048];  // 2 x 4KB double buffer
  int lane = threadIdx.x & 31;
  int wv   = threadIdx.x >> 5;
  int nblk = N >> 6;
  int mblk = blockIdx.x / nblk;
  int nb   = blockIdx.x - mblk * nblk;
  int mt0  = mblk * 16 + wv * 2;
  int ktiles = K >> 5;
  int ntiles = N >> 4;

  const __bf16* abase0 = Ap + ((size_t)mt0 * ktiles) * 512 + lane * 16;
  const __bf16* abase1 = abase0 + (size_t)ktiles * 512;
  // each wave async-copies 512B (32 lanes x 16B) of the block's 4KB B panel
  int chunk = wv * 256 + lane * 8;  // elements

  // issue one async 512B copy of B panel for k-step kt into buffer buf
  auto issue_b = [&](int kt, int buf) {
    const __bf16* gp = Bp + ((size_t)kt * ntiles + nb * 4) * 512 + chunk;
    unsigned long long ga = (unsigned long long)(uintptr_t)gp;
    // generic LDS address truncated to 32 bits == LDS byte offset (ISA 10.2)
    unsigned la = (unsigned)(uintptr_t)&bsh[buf][chunk];
    asm volatile("global_load_async_to_lds_b128 %0, %1, off"
                 :: "v"(la), "v"(ga) : "memory");
  };

  v8f acc[8];
#pragma unroll
  for (int n = 0; n < 8; ++n) acc[n] = (v8f){};

  issue_b(0, 0);
  for (int kt = 0; kt < ktiles; ++kt) {
    int cur = kt & 1;
    if (kt + 1 < ktiles) {
      issue_b(kt + 1, cur ^ 1);
      asm volatile("s_wait_asynccnt 0x1" ::: "memory");  // k-step kt landed
    } else {
      asm volatile("s_wait_asynccnt 0x0" ::: "memory");
    }
    __syncthreads();  // all waves' copies for kt visible

    v16bf a0 = *(const v16bf*)(abase0 + (size_t)kt * 512);
    v16bf a1 = *(const v16bf*)(abase1 + (size_t)kt * 512);
    v16bf b0 = *(const v16bf*)&bsh[cur][0 * 512 + lane * 16];
    v16bf b1 = *(const v16bf*)&bsh[cur][1 * 512 + lane * 16];
    v16bf b2 = *(const v16bf*)&bsh[cur][2 * 512 + lane * 16];
    v16bf b3 = *(const v16bf*)&bsh[cur][3 * 512 + lane * 16];
    acc[0] = WMMA_BF16(a0, b0, acc[0]);
    acc[1] = WMMA_BF16(a0, b1, acc[1]);
    acc[2] = WMMA_BF16(a0, b2, acc[2]);
    acc[3] = WMMA_BF16(a0, b3, acc[3]);
    acc[4] = WMMA_BF16(a1, b0, acc[4]);
    acc[5] = WMMA_BF16(a1, b1, acc[5]);
    acc[6] = WMMA_BF16(a1, b2, acc[6]);
    acc[7] = WMMA_BF16(a1, b3, acc[7]);

    __syncthreads();  // everyone done reading buf `cur` before it is re-filled
  }

  int half = lane >> 4;
  int col0 = (nb << 6) + (lane & 15);
#pragma unroll
  for (int m = 0; m < 2; ++m) {
    int rowb = ((mt0 + m) << 4) + 8 * half;
#pragma unroll
    for (int r = 0; r < 8; ++r) {
      float* crow = C + (size_t)(rowb + r) * N + col0;
      crow[0]  = acc[m * 4 + 0][r];
      crow[16] = acc[m * 4 + 1][r];
      crow[32] = acc[m * 4 + 2][r];
      crow[48] = acc[m * 4 + 3][r];
    }
  }
}

// ---- fused per-head RMS-norm + RoPE (fp32, in place) -----------------------
// grid: (B*S)*nh blocks, 128 threads (one per head dim)
__global__ __launch_bounds__(128) void rmsnorm_rope_kernel(
    float* __restrict__ x, const float* __restrict__ w,
    const float* __restrict__ cosp, const float* __restrict__ sinp, int nh) {
  int bs = blockIdx.x / nh;
  int h  = blockIdx.x - bs * nh;
  int d  = threadIdx.x;  // 0..127
  size_t idx = (size_t)bs * nh * HDc + (size_t)h * HDc + d;
  float v = x[idx];
  float sq = v * v;
#pragma unroll
  for (int m = 16; m >= 1; m >>= 1) sq += __shfl_xor(sq, m, 32);
  __shared__ float red[4];
  if ((threadIdx.x & 31) == 0) red[threadIdx.x >> 5] = sq;
  __syncthreads();
  float tot = red[0] + red[1] + red[2] + red[3];
  float rs  = rsqrtf(tot * (1.0f / HDc) + EPSc);
  float xn  = v * rs * w[d];
  __shared__ float sh[HDc];
  sh[d] = xn;
  __syncthreads();
  float rot = (d < HDc / 2) ? -sh[d + HDc / 2] : sh[d - HDc / 2];
  float c = cosp[(size_t)bs * HDc + d];
  float s = sinp[(size_t)bs * HDc + d];
  x[idx] = xn * c + rot * s;
}

// ---- pack Q (B,S,NH,HD fp32) -> per-head A-layout bf16 ---------------------
__global__ void pack_q_heads_kernel(const float* __restrict__ src, __bf16* __restrict__ dst,
                                    int total) {
  int t = blockIdx.x * blockDim.x + threadIdx.x;
  if (t >= total) return;
  int hh  = t >> 18;        // / (S*HD = 262144)
  int rem = t & 262143;
  int tile = rem >> 9;
  int within = rem & 511;
  int lane = within >> 4, i = within & 15, half = lane >> 4;
  int mt = tile >> 2;       // / (HD/32 = 4)
  int dt = tile & 3;
  int s = (mt << 4) + (lane & 15);
  int d = (dt << 5) + a_kmap(i, half);
  int b = hh / NHc, h = hh - b * NHc;
  dst[t] = (__bf16)src[(size_t)(b * Sc + s) * (NHc * HDc) + h * HDc + d];
}

// ---- pack K (B,S,NKV,HD fp32) -> per-kv-head K^T B-layout bf16 -------------
// tiles: (dt in 0..3) x (st in 0..S/16-1); Kdim=d, N=s
__global__ void pack_kt_heads_kernel(const float* __restrict__ src, __bf16* __restrict__ dst,
                                     int total) {
  int t = blockIdx.x * blockDim.x + threadIdx.x;
  if (t >= total) return;
  int hh  = t >> 18;
  int rem = t & 262143;
  int tile = rem >> 9;
  int within = rem & 511;
  int lane = within >> 4, i = within & 15;
  int dt = tile >> 7;       // / (S/16 = 128)
  int st = tile & 127;
  int d = (dt << 5) + a_kmap(i, lane >> 4);
  int s = (st << 4) + (lane & 15);
  int b = hh / NKVc, kv = hh - b * NKVc;
  dst[t] = (__bf16)src[(size_t)(b * Sc + s) * (NKVc * HDc) + kv * HDc + d];
}

// ---- pack V (B,S,NKV,HD fp32) -> per-kv-head B-layout bf16 -----------------
// tiles: (st in 0..S/32-1) x (dt in 0..7); Kdim=s, N=d
__global__ void pack_v_heads_kernel(const float* __restrict__ src, __bf16* __restrict__ dst,
                                    int total) {
  int t = blockIdx.x * blockDim.x + threadIdx.x;
  if (t >= total) return;
  int hh  = t >> 18;
  int rem = t & 262143;
  int tile = rem >> 9;
  int within = rem & 511;
  int lane = within >> 4, i = within & 15;
  int st = tile >> 3;       // / (HD/16 = 8)
  int dt = tile & 7;
  int s = (st << 5) + a_kmap(i, lane >> 4);
  int d = (dt << 4) + (lane & 15);
  int b = hh / NKVc, kv = hh - b * NKVc;
  dst[t] = (__bf16)src[(size_t)(b * Sc + s) * (NKVc * HDc) + kv * HDc + d];
}

// ---- wave-level causal flash attention -------------------------------------
// One wave handles one (b, h, 16-row q-tile). Q fragments stay resident;
// k-loop streams 32 columns: 2x(4 WMMA) scores, online softmax with 16-lane
// shuffles, P relayout D->A through a 1KB wave-private LDS slice, 8 WMMA
// into the 16x128 fp32 output accumulator.
__global__ __launch_bounds__(256) void flash_attn_kernel(
    const __bf16* __restrict__ Qp, const __bf16* __restrict__ Kp,
    const __bf16* __restrict__ Vp, float* __restrict__ out) {
  __shared__ __attribute__((aligned(32))) __bf16 plds[8][512];
  int lane = threadIdx.x & 31;
  int wv   = threadIdx.x >> 5;
  int w = blockIdx.x * 8 + wv;
  int qt = w & 127;          // S/16 = 128
  int h  = (w >> 7) & 15;
  int b  = w >> 11;
  int kv = h >> 2;           // GQA groups = 4

  const __bf16* qh = Qp + (size_t)(b * NHc + h)  * (Sc * HDc);
  const __bf16* kh = Kp + (size_t)(b * NKVc + kv) * (HDc * Sc);
  const __bf16* vh = Vp + (size_t)(b * NKVc + kv) * (Sc * HDc);

  v16bf qa0 = *(const v16bf*)(qh + ((size_t)qt * 4 + 0) * 512 + lane * 16);
  v16bf qa1 = *(const v16bf*)(qh + ((size_t)qt * 4 + 1) * 512 + lane * 16);
  v16bf qa2 = *(const v16bf*)(qh + ((size_t)qt * 4 + 2) * 512 + lane * 16);
  v16bf qa3 = *(const v16bf*)(qh + ((size_t)qt * 4 + 3) * 512 + lane * 16);

  v8f acc[8];
#pragma unroll
  for (int n = 0; n < 8; ++n) acc[n] = (v8f){};
  float mprev[8], lsum[8];
#pragma unroll
  for (int r = 0; r < 8; ++r) { mprev[r] = -__builtin_inff(); lsum[r] = 0.0f; }

  int half = lane >> 4;
  int lcol = lane & 15;
  const float scale = 0.08838834764831845f;  // 1/sqrt(128)
  int qrow_base = qt * 16 + 8 * half;
  int nkb = (qt + 2) >> 1;   // ceil((qt*16+16)/32)

  for (int kb = 0; kb < nkb; ++kb) {
    float p0[8], p1[8];
    // ---- scores: two 16x16 tiles covering 32 key columns
#pragma unroll
    for (int c = 0; c < 2; ++c) {
      int st = kb * 2 + c;
      v8f sf = (v8f){};
      v16bf kb0 = *(const v16bf*)(kh + ((size_t)0 * 128 + st) * 512 + lane * 16);
      v16bf kb1 = *(const v16bf*)(kh + ((size_t)1 * 128 + st) * 512 + lane * 16);
      v16bf kb2 = *(const v16bf*)(kh + ((size_t)2 * 128 + st) * 512 + lane * 16);
      v16bf kb3 = *(const v16bf*)(kh + ((size_t)3 * 128 + st) * 512 + lane * 16);
      sf = WMMA_BF16(qa0, kb0, sf);
      sf = WMMA_BF16(qa1, kb1, sf);
      sf = WMMA_BF16(qa2, kb2, sf);
      sf = WMMA_BF16(qa3, kb3, sf);
      int col = st * 16 + lcol;
#pragma unroll
      for (int r = 0; r < 8; ++r) {
        int row = qrow_base + r;
        float s = sf[r] * scale + ((col <= row) ? 0.0f : -1e9f);
        if (c == 0) p0[r] = s; else p1[r] = s;
      }
    }
    // ---- online softmax (row stats live across 16-lane groups)
    float sfac[8];
#pragma unroll
    for (int r = 0; r < 8; ++r) {
      float bm = fmaxf(p0[r], p1[r]);
#pragma unroll
      for (int m = 8; m >= 1; m >>= 1) bm = fmaxf(bm, __shfl_xor(bm, m, 16));
      float mnew = fmaxf(mprev[r], bm);
      float e0 = __expf(p0[r] - mnew);
      float e1 = __expf(p1[r] - mnew);
      float rs = e0 + e1;
#pragma unroll
      for (int m = 8; m >= 1; m >>= 1) rs += __shfl_xor(rs, m, 16);
      float f = __expf(mprev[r] - mnew);
      lsum[r] = lsum[r] * f + rs;
      mprev[r] = mnew;
      sfac[r] = f;
      p0[r] = e0;
      p1[r] = e1;
    }
#pragma unroll
    for (int n = 0; n < 8; ++n)
#pragma unroll
      for (int r = 0; r < 8; ++r) acc[n][r] *= sfac[r];

    // ---- relayout P: D-frag (f32) -> A-frag (bf16) via wave-private LDS
#pragma unroll
    for (int c = 0; c < 2; ++c) {
#pragma unroll
      for (int r = 0; r < 8; ++r) {
        int row_in = r + 8 * half;
        int col_in = c * 16 + lcol;
        int hs, is;
        if (col_in < 8)       { hs = 0; is = col_in;      }
        else if (col_in < 16) { hs = 1; is = col_in - 8;  }
        else if (col_in < 24) { hs = 0; is = col_in - 8;  }
        else                  { hs = 1; is = col_in - 16; }
        plds[wv][(row_in + 16 * hs) * 16 + is] = (__bf16)(c ? p1[r] : p0[r]);
      }
    }
    asm volatile("s_wait_dscnt 0x0" ::: "memory");  // wave-private; DS is in-order per wave
    v16bf pa = *(const v16bf*)(&plds[wv][lane * 16]);

    // ---- P @ V: 8 WMMAs over the 128-wide head dim
#pragma unroll
    for (int n = 0; n < 8; ++n) {
      v16bf bv = *(const v16bf*)(vh + ((size_t)kb * 8 + n) * 512 + lane * 16);
      acc[n] = WMMA_BF16(pa, bv, acc[n]);
    }
  }

  // ---- normalize and write fp32 attention output (b*S+s, h*HD+d)
#pragma unroll
  for (int r = 0; r < 8; ++r) {
    float inv = 1.0f / lsum[r];
    size_t row = (size_t)b * Sc + qt * 16 + r + 8 * half;
#pragma unroll
    for (int n = 0; n < 8; ++n) {
      out[row * (NHc * HDc) + h * HDc + n * 16 + lcol] = acc[n][r] * inv;
    }
  }
}

// ===========================================================================
// Host-side orchestration
// ===========================================================================
extern "C" void kernel_launch(void* const* d_in, const int* in_sizes, int n_in,
                              void* d_out, int out_size, void* d_ws, size_t ws_size,
                              hipStream_t stream) {
  const float* X       = (const float*)d_in[0];  // (B,S,HID)
  const float* cosp    = (const float*)d_in[1];  // (B,S,HD)
  const float* sinp    = (const float*)d_in[2];  // (B,S,HD)
  // d_in[3] = attention_mask (causal, applied analytically)
  const float* Wq      = (const float*)d_in[4];  // (HID, NH*HD)
  const float* Wk      = (const float*)d_in[5];  // (HID, NKV*HD)
  const float* Wv      = (const float*)d_in[6];  // (HID, NKV*HD)
  const float* Wo      = (const float*)d_in[7];  // (NH*HD, HID)
  const float* qnw     = (const float*)d_in[8];  // (HD,)
  const float* knw     = (const float*)d_in[9];  // (HD,)
  float* out = (float*)d_out;

  char* ws = (char*)d_ws;
  const size_t MBy = 1ull << 20;
  __bf16* Wqp  = (__bf16*)(ws + 0);             //  8 MB  (2048x2048 bf16)
  __bf16* Wkp  = (__bf16*)(ws + 8  * MBy);      //  2 MB
  __bf16* Wvp  = (__bf16*)(ws + 10 * MBy);      //  2 MB
  __bf16* Wop  = (__bf16*)(ws + 12 * MBy);      //  8 MB
  __bf16* Xp   = (__bf16*)(ws + 20 * MBy);      // 16 MB  (also reused for packed attn)
  float*  Qr   = (float*) (ws + 36 * MBy);      // 32 MB  (also reused for attn out)
  float*  Kr   = (float*) (ws + 68 * MBy);      //  8 MB
  float*  Vr   = (float*) (ws + 76 * MBy);      //  8 MB
  __bf16* Qp   = (__bf16*)(ws + 84 * MBy);      // 16 MB
  __bf16* Kp   = (__bf16*)(ws + 100 * MBy);     //  4 MB
  __bf16* Vp   = (__bf16*)(ws + 104 * MBy);     //  4 MB  (end = 108 MB)

  const int BS = Bc * Sc;                 // 4096
  const int NQ = NHc * HDc;               // 2048
  const int NK = NKVc * HDc;              // 512

  // 1) pack weights into B-layout bf16
  pack_b_kernel<<<(HIDc * NQ) / 256, 256, 0, stream>>>(Wq, Wqp, HIDc, NQ, HIDc * NQ);
  pack_b_kernel<<<(HIDc * NK) / 256, 256, 0, stream>>>(Wk, Wkp, HIDc, NK, HIDc * NK);
  pack_b_kernel<<<(HIDc * NK) / 256, 256, 0, stream>>>(Wv, Wvp, HIDc, NK, HIDc * NK);
  pack_b_kernel<<<(NQ * HIDc) / 256, 256, 0, stream>>>(Wo, Wop, NQ, HIDc, NQ * HIDc);

  // 2) pack activations into A-layout bf16
  pack_a_kernel<<<(BS * HIDc) / 256, 256, 0, stream>>>(X, Xp, BS, HIDc, BS * HIDc);

  // 3) QKV projections (bf16 WMMA, fp32 out); grid = (M/256)*(N/64) blocks
  gemm_bf16_kernel<<<(BS / 256) * (NQ / 64), 256, 0, stream>>>(Xp, Wqp, Qr, BS, HIDc, NQ);
  gemm_bf16_kernel<<<(BS / 256) * (NK / 64), 256, 0, stream>>>(Xp, Wkp, Kr, BS, HIDc, NK);
  gemm_bf16_kernel<<<(BS / 256) * (NK / 64), 256, 0, stream>>>(Xp, Wvp, Vr, BS, HIDc, NK);

  // 4) per-head RMS-norm + RoPE (in place, fp32)
  rmsnorm_rope_kernel<<<BS * NHc,  128, 0, stream>>>(Qr, qnw, cosp, sinp, NHc);
  rmsnorm_rope_kernel<<<BS * NKVc, 128, 0, stream>>>(Kr, knw, cosp, sinp, NKVc);

  // 5) pack Q / K^T / V into per-head WMMA fragment layouts
  pack_q_heads_kernel<<<(Bc * NHc  * Sc * HDc) / 256, 256, 0, stream>>>(Qr, Qp, Bc * NHc  * Sc * HDc);
  pack_kt_heads_kernel<<<(Bc * NKVc * Sc * HDc) / 256, 256, 0, stream>>>(Kr, Kp, Bc * NKVc * Sc * HDc);
  pack_v_heads_kernel<<<(Bc * NKVc * Sc * HDc) / 256, 256, 0, stream>>>(Vr, Vp, Bc * NKVc * Sc * HDc);

  // 6) causal flash attention (reuses Qr buffer for fp32 attn output)
  float* attn = Qr;
  flash_attn_kernel<<<(Bc * NHc * (Sc / 16)) / 8, 256, 0, stream>>>(Qp, Kp, Vp, attn);

  // 7) output projection (reuses Xp buffer for packed attn)
  pack_a_kernel<<<(BS * NQ) / 256, 256, 0, stream>>>(attn, Xp, BS, NQ, BS * NQ);
  gemm_bf16_kernel<<<(BS / 256) * (HIDc / 64), 256, 0, stream>>>(Xp, Wop, out, BS, NQ, HIDc);
}
